// MultiHeadLatentAttention_70265664963016
// MI455X (gfx1250) — matrix operand order repaired and checked by reading, not compiled, not run on hardware
//
#include <hip/hip_runtime.h>
#include <cstdint>
#include <cstddef>

// ---------- types ----------
typedef __attribute__((ext_vector_type(16))) __bf16    v16bf;
typedef __attribute__((ext_vector_type(8)))  __bf16    v8bf;
typedef __attribute__((ext_vector_type(8)))  float     v8f;
typedef __attribute__((ext_vector_type(4)))  uint32_t  v4u;
typedef __attribute__((ext_vector_type(8)))  int       v8i;
typedef __attribute__((ext_vector_type(4)))  int       v4i;

static __device__ __forceinline__ __bf16 f2bf(float f) {
    union { float f; uint32_t u; } v; v.f = f;
    uint32_t r = (v.u + 0x7FFFu + ((v.u >> 16) & 1u)) >> 16;
    union { unsigned short s; __bf16 b; } o; o.s = (unsigned short)r;
    return o.b;
}

static __device__ __forceinline__ v8f wmma_bf16(v16bf a, v16bf b, v8f c) {
    return __builtin_amdgcn_wmma_f32_16x16x32_bf16(
        /*neg_a=*/false, a, /*neg_b=*/false, b,
        /*c_mod=*/(short)0, c, /*reuse_a=*/false, /*reuse_b=*/false);
}

// A-matrix 16x32 bf16 fragment (this code's contiguous K order, same as B): lane holds row M=lane&15;
// K halves: lanes 0-15 -> K {0..15}; lanes 16-31 -> K {16..31}.
static __device__ __forceinline__ v16bf load_fragA(const __bf16* row_kk, int c0) {
    union { v16bf v; v8bf h[2]; } u;
    u.h[0] = *(const v8bf*)(row_kk + 2 * c0);
    u.h[1] = *(const v8bf*)(row_kk + 2 * c0 + 8);
    return u.v;
}

// Generic-pointer low 32 bits == LDS byte offset (ISA 10.2 aperture mapping).
static __device__ __forceinline__ uint32_t lds_off(const void* p) {
    return (uint32_t)(uintptr_t)p;
}

// ---------- TDM: stage a rows x cols (bf16) tile, row stride = stride elems ----------
// D# per CDNA5 ISA. tensor dims == tile dims (never OOB here).
// 6-arg builtin form (clang-23 lane): (v4u g0, v8i g1, v4i g2, v4i g3, v8i g4, i32 cpol)
static __device__ __forceinline__ void tdm_load_2d(const __bf16* gsrc, uint32_t ldsaddr,
                                                   uint32_t rows, uint32_t cols,
                                                   uint32_t stride_elems)
{
    uint64_t ga = (uint64_t)(uintptr_t)gsrc;
    v4u g0;
    g0[0] = 1u;                                   // count=1, user descriptor, no gather
    g0[1] = ldsaddr;                              // lds_addr (bytes)
    g0[2] = (uint32_t)ga;                         // global_addr[31:0]
    g0[3] = (uint32_t)((ga >> 32) & 0x01FFFFFFu)  // global_addr[56:32]
          | (2u << 30);                           // type = 2 ("image")
    uint64_t st = (uint64_t)stride_elems;         // tensor_dim0_stride (48b)
    v8i g1;
    g1[0] = (int)(1u << 16);                                              // data_size=1 (2B)
    g1[1] = (int)((cols & 0xFFFFu) << 16);                                // tensor_dim0 lo
    g1[2] = (int)(((cols >> 16) & 0xFFFFu) | ((rows & 0xFFFFu) << 16));   // dim0 hi | dim1 lo
    g1[3] = (int)(((rows >> 16) & 0xFFFFu) | ((cols & 0xFFFFu) << 16));   // dim1 hi | tile_dim0
    g1[4] = (int)(rows & 0xFFFFu);                                        // tile_dim1 (tile_dim2=0)
    g1[5] = (int)(uint32_t)(st & 0xFFFFFFFFu);                            // stride[31:0]
    g1[6] = (int)(uint32_t)((st >> 32) & 0xFFFFu);                        // stride[47:32]
    g1[7] = 0;
    v4i z4 = {};
    v8i z8 = {};
    __builtin_amdgcn_tensor_load_to_lds(g0, g1, z4, z4, z8, 0);
}

// ---------- fp32 -> bf16 convert ----------
__global__ void cvt_f32_bf16(const float* __restrict__ in, __bf16* __restrict__ out, size_t n) {
    size_t i = (size_t)blockIdx.x * blockDim.x + threadIdx.x;
    size_t stride = (size_t)gridDim.x * blockDim.x;
    for (; i < n; i += stride) out[i] = f2bf(in[i]);
}

// ---------- WMMA GEMM with TDM-staged LDS tiles: C(M,N) = A(M,K) @ W(N,K)^T ----------
// 128 threads = 4 waves (2x2); block tile 64 x BN; wave tile 32 x BN/2.
// Double-buffered: wave 0 issues tensor_load_to_lds for the next K-tile while
// all waves compute the current one from LDS. Requires M%64==0, N%BN==0, K%32==0.
template<int BN>
__global__ __launch_bounds__(128)
void gemm_bf16(const __bf16* __restrict__ A, const __bf16* __restrict__ W,
               float* __restrict__ Cf, __bf16* __restrict__ Cb,
               const float* __restrict__ bias, int M, int N, int K)
{
    constexpr int WN = BN / 2;
    constexpr int NF = WN / 16;     // B fragments per wave (1 or 2)

    __shared__ __attribute__((aligned(64))) __bf16 sA[2][64][32];
    __shared__ __attribute__((aligned(64))) __bf16 sW[2][BN][32];

    const int lane = threadIdx.x & 31;
    const int wave = threadIdx.x >> 5;
    const int wm   = wave >> 1;
    const int wn   = wave & 1;
    const int m0   = blockIdx.y * 64;
    const int n0   = blockIdx.x * BN;
    const int lr   = lane & 15;
    const int lk   = (lane >> 4) * 16;   // B-fragment K offset
    const int c0   = (lane >> 4) * 8;    // A-fragment K sub-offset

    v8f c[2][NF] = {};

    if (wave == 0) {   // prologue: stage K-tile 0
        tdm_load_2d(A + (size_t)m0 * K, lds_off(&sA[0][0][0]), 64, 32, (uint32_t)K);
        tdm_load_2d(W + (size_t)n0 * K, lds_off(&sW[0][0][0]), BN, 32, (uint32_t)K);
    }

    int buf = 0;
    for (int kk = 0; kk < K; kk += 32) {
        if (wave == 0) {
            if (kk + 32 < K) {   // prefetch next K-tile into the other buffer
                tdm_load_2d(A + (size_t)m0 * K + kk + 32, lds_off(&sA[buf ^ 1][0][0]), 64, 32, (uint32_t)K);
                tdm_load_2d(W + (size_t)n0 * K + kk + 32, lds_off(&sW[buf ^ 1][0][0]), BN, 32, (uint32_t)K);
                __builtin_amdgcn_s_wait_tensorcnt(2);   // TDM in-order: current pair done
            } else {
                __builtin_amdgcn_s_wait_tensorcnt(0);
            }
        }
        __syncthreads();   // current buffer visible to all waves

        v16bf a0 = load_fragA(&sA[buf][wm * 32 + lr][0], c0);
        v16bf a1 = load_fragA(&sA[buf][wm * 32 + 16 + lr][0], c0);
        v16bf b[NF];
#pragma unroll
        for (int j = 0; j < NF; ++j)
            b[j] = *(const v16bf*)(&sW[buf][wn * WN + j * 16 + lr][lk]);
#pragma unroll
        for (int j = 0; j < NF; ++j) {
            c[0][j] = wmma_bf16(a0, b[j], c[0][j]);
            c[1][j] = wmma_bf16(a1, b[j], c[1][j]);
        }

        __syncthreads();   // all reads done before buffer is overwritten
        buf ^= 1;
    }

    const int rowh = (lane >> 4) * 8;
#pragma unroll
    for (int i = 0; i < 2; ++i) {
#pragma unroll
        for (int j = 0; j < NF; ++j) {
#pragma unroll
            for (int r = 0; r < 8; ++r) {
                int m = m0 + wm * 32 + i * 16 + rowh + r;
                int n = n0 + wn * WN + j * 16 + lr;
                float v = c[i][j][r];
                if (bias) v += bias[n];
                if (Cf) Cf[(size_t)m * N + n] = v;
                if (Cb) Cb[(size_t)m * N + n] = f2bf(v);
            }
        }
    }
}

// ---------- assemble q_t, k_t (RoPE) and v_t (transposed) ----------
// q_t,k_t: (B,H,S,128) bf16 ; v_t: (B,H,128,S) bf16
__global__ __launch_bounds__(128)
void assemble_qkv(const __bf16* __restrict__ qc,  // (B*S, 1536), col = h*96 + c
                  const float*  __restrict__ qr,  // (B*S, 512),  col = h*32 + j
                  const __bf16* __restrict__ kc,  // (B*S, 1536)
                  const float*  __restrict__ kr,  // (B*S, 32)
                  const __bf16* __restrict__ vb,  // (B*S, 2048), col = h*128 + d
                  __bf16* __restrict__ qt, __bf16* __restrict__ kt, __bf16* __restrict__ vt)
{
    const int S = 2048, H = 16;
    const int d = threadIdx.x;
    const int s = blockIdx.x;
    const int h = blockIdx.y;
    const int b = blockIdx.z;
    const size_t tok = (size_t)b * S + s;

    // v transpose: v_t[b][h][d][s]
    vt[(((size_t)b * H + h) * 128 + d) * S + s] = vb[tok * 2048 + h * 128 + d];

    __bf16 qv, kv;
    if (d < 96) {
        qv = qc[tok * 1536 + h * 96 + d];
        kv = kc[tok * 1536 + h * 96 + d];
    } else {
        int j = d - 96;                 // 0..31 (rope dim R=32)
        int i = j & 15;
        float freq = __powf(10000.0f, -(float)i / 16.0f);   // inv_freq
        float ang  = (float)s * freq;
        float cs = __cosf(ang), sn = __sinf(ang);
        float q  = qr[tok * 512 + h * 32 + j];
        float qrot = (j < 16) ? -qr[tok * 512 + h * 32 + j + 16]
                              :  qr[tok * 512 + h * 32 + j - 16];
        qv = f2bf(q * cs + qrot * sn);
        float k  = kr[tok * 32 + j];
        float krot = (j < 16) ? -kr[tok * 32 + j + 16]
                              :  kr[tok * 32 + j - 16];
        kv = f2bf(k * cs + krot * sn);
    }
    qt[(((size_t)b * H + h) * S + s) * 128 + d] = qv;
    kt[(((size_t)b * H + h) * S + s) * 128 + d] = kv;
}

// ---------- causal flash attention: 1 wave handles 16 q-rows of one (b,h) ----------
// qt,kt: (B,H,S,128) bf16 ; vt: (B,H,128,S) bf16 ; ctx: (B*S, 2048) bf16 (col = h*128+d)
__global__ __launch_bounds__(32)
void mla_flash_attn(const __bf16* __restrict__ qt, const __bf16* __restrict__ kt,
                    const __bf16* __restrict__ vt, __bf16* __restrict__ ctx)
{
    const int S = 2048, H = 16, DH = 128;
    const float scale = 0.088388347648318447f;   // 1/sqrt(128)

    const int q0 = blockIdx.x * 16;
    const int h  = blockIdx.y;
    const int b  = blockIdx.z;
    const int lane = threadIdx.x;
    const int lr = lane & 15;
    const int lk = (lane >> 4) * 16;
    const int c0 = (lane >> 4) * 8;
    const int rowh = (lane >> 4) * 8;

    const __bf16* qbase = qt + ((size_t)b * H + h) * (size_t)S * DH;
    const __bf16* kbase = kt + ((size_t)b * H + h) * (size_t)S * DH;
    const __bf16* vbase = vt + ((size_t)b * H + h) * (size_t)DH * S;

    __shared__ __attribute__((aligned(64))) __bf16 plds[16][32];

    // Q fragments: 128 = 4 K-steps of 32
    v16bf qa[4];
    const __bf16* qrow = qbase + (size_t)(q0 + lr) * DH;
#pragma unroll
    for (int d = 0; d < 4; ++d) qa[d] = load_fragA(qrow + d * 32, c0);

    v8f acc[8] = {};               // 16 x 128 fp32 context accumulator
    float mi[8], li[8];
#pragma unroll
    for (int r = 0; r < 8; ++r) { mi[r] = -3.0e38f; li[r] = 0.0f; }

    const int kbmax = (q0 + 15) >> 5;
    for (int kb = 0; kb <= kbmax; ++kb) {
        const int kpos = kb * 32;

        if (kb < kbmax)  // L2 warm-up for next K tile (global_prefetch)
            __builtin_prefetch(kbase + (size_t)(kpos + 32) * DH, 0, 1);

        // scores: 16 x 32 as two 16x16 WMMA tiles
        v8f s0 = {}, s1 = {};
#pragma unroll
        for (int d = 0; d < 4; ++d) {
            v16bf k0 = *(const v16bf*)(kbase + (size_t)(kpos + lr) * DH + d * 32 + lk);
            v16bf k1 = *(const v16bf*)(kbase + (size_t)(kpos + 16 + lr) * DH + d * 32 + lk);
            s0 = wmma_bf16(qa[d], k0, s0);
            s1 = wmma_bf16(qa[d], k1, s1);
        }
#pragma unroll
        for (int r = 0; r < 8; ++r) { s0[r] *= scale; s1[r] *= scale; }

        if (kpos + 31 > q0) {   // diagonal-overlapping block: causal mask
#pragma unroll
            for (int r = 0; r < 8; ++r) {
                int row = q0 + rowh + r;
                if (kpos + lr      > row) s0[r] = -10000.0f;
                if (kpos + 16 + lr > row) s1[r] = -10000.0f;
            }
        }

        // online softmax (row r lives in a 16-lane half: reduce with shfl_xor)
#pragma unroll
        for (int r = 0; r < 8; ++r) {
            float t = fmaxf(s0[r], s1[r]);
            t = fmaxf(t, __shfl_xor(t, 1));
            t = fmaxf(t, __shfl_xor(t, 2));
            t = fmaxf(t, __shfl_xor(t, 4));
            t = fmaxf(t, __shfl_xor(t, 8));
            float mn = fmaxf(mi[r], t);
            float sc = __expf(mi[r] - mn);
            mi[r] = mn;
            li[r] *= sc;
#pragma unroll
            for (int j = 0; j < 8; ++j) acc[j][r] *= sc;
            float p0 = __expf(s0[r] - mn);
            float p1 = __expf(s1[r] - mn);
            float ps = p0 + p1;
            ps += __shfl_xor(ps, 1);
            ps += __shfl_xor(ps, 2);
            ps += __shfl_xor(ps, 4);
            ps += __shfl_xor(ps, 8);
            li[r] += ps;
            plds[rowh + r][lr]      = f2bf(p0);
            plds[rowh + r][16 + lr] = f2bf(p1);
        }
        __syncthreads();
        // P as A-fragment (16x32) from LDS
        union { v16bf v; v8bf hh[2]; } pu;
        pu.hh[0] = *(const v8bf*)(&plds[lr][2 * c0]);
        pu.hh[1] = *(const v8bf*)(&plds[lr][2 * c0 + 8]);
        v16bf pa = pu.v;
        __syncthreads();

        // ctx += P @ V ; V fragments contiguous thanks to (b,h,d,S) layout
#pragma unroll
        for (int j = 0; j < 8; ++j) {
            v16bf vb = *(const v16bf*)(vbase + (size_t)(j * 16 + lr) * S + kpos + lk);
            acc[j] = wmma_bf16(pa, vb, acc[j]);
        }
    }

    // epilogue: normalize and store ctx (b, s, h*128 + d) bf16
#pragma unroll
    for (int r = 0; r < 8; ++r) {
        float inv = 1.0f / li[r];
        size_t rowoff = ((size_t)b * S + q0 + rowh + r) * 2048 + (size_t)h * 128;
#pragma unroll
        for (int j = 0; j < 8; ++j)
            ctx[rowoff + j * 16 + lr] = f2bf(acc[j][r] * inv);
    }
}

// ---------- host launcher ----------
extern "C" void kernel_launch(void* const* d_in, const int* in_sizes, int n_in,
                              void* d_out, int out_size, void* d_ws, size_t ws_size,
                              hipStream_t stream)
{
    const int B = 2, S = 2048, HID = 2048, H = 16, LAT = 512, R = 32, DH = 128, C = 96;
    const int M = B * S;   // 4096 tokens

    const float* x       = (const float*)d_in[0];
    const float* Wq_down = (const float*)d_in[1];   // (512, 2048)
    const float* Wq_up   = (const float*)d_in[2];   // (1536, 512)
    const float* Wq_rope = (const float*)d_in[3];   // (512, 512)
    const float* Wkv_down= (const float*)d_in[4];   // (512, 2048)
    const float* Wk_up   = (const float*)d_in[5];   // (1536, 512)
    const float* Wk_rope = (const float*)d_in[6];   // (32, 2048)
    const float* Wv_up   = (const float*)d_in[7];   // (2048, 512)
    const float* Wo      = (const float*)d_in[8];   // (2048, 2048)
    const float* bo      = (const float*)d_in[9];   // (2048)

    char* ws = (char*)d_ws;
    size_t off = 0;
    auto alloc = [&](size_t bytes) -> void* {
        void* p = ws + off;
        off += (bytes + 255) & ~(size_t)255;
        return p;
    };

    // bf16 copies of inputs
    __bf16* xb    = (__bf16*)alloc((size_t)M * HID * 2);
    __bf16* wqd_b = (__bf16*)alloc((size_t)LAT * HID * 2);
    __bf16* wqu_b = (__bf16*)alloc((size_t)C * H * LAT * 2);
    __bf16* wqr_b = (__bf16*)alloc((size_t)R * H * LAT * 2);
    __bf16* wkd_b = (__bf16*)alloc((size_t)LAT * HID * 2);
    __bf16* wku_b = (__bf16*)alloc((size_t)C * H * LAT * 2);
    __bf16* wkr_b = (__bf16*)alloc((size_t)R * HID * 2);
    __bf16* wvu_b = (__bf16*)alloc((size_t)HID * LAT * 2);
    __bf16* wo_b  = (__bf16*)alloc((size_t)HID * HID * 2);
    // intermediates
    __bf16* qlat_b = (__bf16*)alloc((size_t)M * LAT * 2);
    __bf16* kvlat_b= (__bf16*)alloc((size_t)M * LAT * 2);
    __bf16* qc_b   = (__bf16*)alloc((size_t)M * C * H * 2);
    float*  qr_f   = (float*) alloc((size_t)M * R * H * 4);
    __bf16* kc_b   = (__bf16*)alloc((size_t)M * C * H * 2);
    float*  kr_f   = (float*) alloc((size_t)M * R * 4);
    __bf16* v_b    = (__bf16*)alloc((size_t)M * HID * 2);
    __bf16* qt     = (__bf16*)alloc((size_t)B * H * S * DH * 2);
    __bf16* kt     = (__bf16*)alloc((size_t)B * H * S * DH * 2);
    __bf16* vt     = (__bf16*)alloc((size_t)B * H * DH * S * 2);
    __bf16* ctx_b  = (__bf16*)alloc((size_t)M * HID * 2);

    // 1) convert fp32 -> bf16
    const float* csrc[9] = { x, Wq_down, Wq_up, Wq_rope, Wkv_down, Wk_up, Wk_rope, Wv_up, Wo };
    __bf16* cdst[9] = { xb, wqd_b, wqu_b, wqr_b, wkd_b, wku_b, wkr_b, wvu_b, wo_b };
    size_t cn[9] = { (size_t)M*HID, (size_t)LAT*HID, (size_t)C*H*LAT, (size_t)R*H*LAT,
                     (size_t)LAT*HID, (size_t)C*H*LAT, (size_t)R*HID, (size_t)HID*LAT,
                     (size_t)HID*HID };
    for (int i = 0; i < 9; ++i)
        cvt_f32_bf16<<<1024, 256, 0, stream>>>(csrc[i], cdst[i], cn[i]);

    dim3 blk(128);
    // 2) down projections: q_lat, kv_lat  (4096 x 512 = x @ W^T, K=2048)
    gemm_bf16<64><<<dim3(LAT/64, M/64), blk, 0, stream>>>(xb, wqd_b, nullptr, qlat_b, nullptr, M, LAT, HID);
    gemm_bf16<64><<<dim3(LAT/64, M/64), blk, 0, stream>>>(xb, wkd_b, nullptr, kvlat_b, nullptr, M, LAT, HID);
    // 3) up projections
    gemm_bf16<64><<<dim3((C*H)/64, M/64), blk, 0, stream>>>(qlat_b, wqu_b, nullptr, qc_b, nullptr, M, C*H, LAT);
    gemm_bf16<64><<<dim3((R*H)/64, M/64), blk, 0, stream>>>(qlat_b, wqr_b, qr_f, nullptr, nullptr, M, R*H, LAT);
    gemm_bf16<64><<<dim3((C*H)/64, M/64), blk, 0, stream>>>(kvlat_b, wku_b, nullptr, kc_b, nullptr, M, C*H, LAT);
    gemm_bf16<32><<<dim3(R/32, M/64),   blk, 0, stream>>>(xb, wkr_b, kr_f, nullptr, nullptr, M, R, HID);
    gemm_bf16<64><<<dim3(HID/64, M/64), blk, 0, stream>>>(kvlat_b, wvu_b, nullptr, v_b, nullptr, M, HID, LAT);
    // 4) RoPE + head-major staging (+ V transpose for contiguous P@V fragments)
    assemble_qkv<<<dim3(S, H, B), 128, 0, stream>>>(qc_b, qr_f, kc_b, kr_f, v_b, qt, kt, vt);
    // 5) causal flash attention (1 wave per 16 q-rows per head)
    mla_flash_attn<<<dim3(S/16, H, B), 32, 0, stream>>>(qt, kt, vt, ctx_b);
    // 6) output projection with bias -> fp32
    gemm_bf16<64><<<dim3(HID/64, M/64), blk, 0, stream>>>(ctx_b, wo_b, (float*)d_out, nullptr, bo, M, HID, HID);
}
